// AttentionPooling_83494164234742
// MI455X (gfx1250) — matrix operand order repaired and checked by reading, not compiled, not run on hardware
//
#include <hip/hip_runtime.h>
#include <math.h>

#define B_SZ 4096
#define T_SZ 200
#define E_SZ 64
#define H1_SZ 80
#define H2_SZ 40
#define F_SZ 256

#define WAVES 4
#define BLOCK (WAVES * 32)
#define MTILES 13      // ceil(200 / 16)

#define KB1 8          // F / 32 K-steps for layer 1
#define NT1 5          // H1 / 16 N-tiles for layer 1
#define K2P 96         // layer-2 K padded (80 -> 96)
#define KB2 3
#define N2P 48         // layer-2 N padded (40 -> 48)
#define NT2 3

typedef __attribute__((ext_vector_type(16))) __bf16        v16bf;
typedef __attribute__((ext_vector_type(4)))  __bf16        bf4v;
typedef __attribute__((ext_vector_type(2)))  __bf16        bf2v;
typedef __attribute__((ext_vector_type(8)))  float          v8f;
typedef __attribute__((ext_vector_type(8)))  unsigned short us8;
typedef __attribute__((ext_vector_type(16))) unsigned short us16;

// pack 4 floats -> 4 bf16 (v_cvt_pk_bf16_f32), one 8B LDS store
__device__ __forceinline__ void store_bf4(unsigned short* p, float a, float b, float c, float d) {
  bf4v v = { (__bf16)a, (__bf16)b, (__bf16)c, (__bf16)d };
  *(bf4v*)p = v;
}
__device__ __forceinline__ void store_bf1(unsigned short* p, float x) {
  *(__bf16*)p = (__bf16)x;
}
__device__ __forceinline__ float load_bf1(const unsigned short* p) {
  return (float)(*(const __bf16*)p);
}

// B-fragment: each lane's 16 bf16 stored contiguously (32B) at staging time.
__device__ __forceinline__ v16bf load_b_frag(const unsigned short* p) {
  return __builtin_bit_cast(v16bf, *(const us16*)p);
}

// A-fragment from row-major bf16 LDS tile, contiguous-K order matching the B frags:
// lane l: row M = l&15; elements = K[kBase + koff .. +16) contiguous
// with koff = (l>>4)*16.  Both chunks are 16B-aligned for our row strides.
__device__ __forceinline__ v16bf load_a_frag(const unsigned short* base, int rowStride,
                                             int kBase, int lane) {
  const int row  = lane & 15;
  const int koff = (lane >> 4) << 4;
  const unsigned short* p0 = base + row * rowStride + kBase + koff;
  us8 c0 = *(const us8*)p0;
  us8 c1 = *(const us8*)(p0 + 8);
  us16 u = __builtin_shufflevector(c0, c1, 0,1,2,3,4,5,6,7,8,9,10,11,12,13,14,15);
  return __builtin_bit_cast(v16bf, u);
}

__global__ __launch_bounds__(BLOCK, 1)
void attnpool_wmma_kernel(const float* __restrict__ queries,
                          const float* __restrict__ keys,
                          const int*   __restrict__ key_masks,
                          const float* __restrict__ pos_q,
                          const float* __restrict__ pos_k,
                          const float* __restrict__ W1, const float* __restrict__ b1,
                          const float* __restrict__ W2, const float* __restrict__ b2,
                          const float* __restrict__ W3, const float* __restrict__ b3,
                          float* __restrict__ out)
{
  // Weights staged once per block, pre-swizzled to B-fragment layout:
  // element e of lane l of frag(kb,nt):  N = nt*16 + (l&15),  K = kb*32 + (l>>4)*16 + e
  __shared__ __align__(32) unsigned short sW1f[KB1 * NT1 * 512];  // 40 KB
  __shared__ __align__(32) unsigned short sW2f[KB2 * NT2 * 512];  //  9 KB
  __shared__ float sW3[H2_SZ];
  __shared__ float sB1[H1_SZ];
  __shared__ float sB2[N2P];
  __shared__ __align__(16) float sQ[E_SZ];
  __shared__ float sOut[E_SZ];
  // Per-wave scratch
  __shared__ __align__(32) unsigned short sAtt[WAVES * 16 * F_SZ]; // att_in tiles (32 KB)
  __shared__ __align__(32) unsigned short sHa [WAVES * 16 * K2P];  // layer-1 out (12 KB)
  __shared__ __align__(32) unsigned short sHb [WAVES * 16 * N2P];  // layer-2 out (6 KB)
  __shared__ float sWv[WAVES * 16];                                // per-row weights

  const int tid = threadIdx.x;
  const int b   = blockIdx.x;

  // ---------------- Phase 0: stage weights / query ----------------
  // W1: vectorized read (float4), scattered bf16 stores into fragment layout.
  // A float4 covers cols n0..n0+3 of row k (16 | 4, so it stays inside one N-tile).
  for (int i4 = tid; i4 < (F_SZ * H1_SZ) / 4; i4 += BLOCK) {
    int k  = i4 / (H1_SZ / 4);
    int n0 = (i4 % (H1_SZ / 4)) * 4;
    float4 w = *(const float4*)(W1 + k * H1_SZ + n0);
    int kb = k >> 5, lhi = (k >> 4) & 1, e = k & 15;
    int nt = n0 >> 4, nlo = n0 & 15;
    unsigned short* fp = &sW1f[(kb * NT1 + nt) * 512 + e];
    store_bf1(&fp[(lhi * 16 + nlo + 0) * 16], w.x);
    store_bf1(&fp[(lhi * 16 + nlo + 1) * 16], w.y);
    store_bf1(&fp[(lhi * 16 + nlo + 2) * 16], w.z);
    store_bf1(&fp[(lhi * 16 + nlo + 3) * 16], w.w);
  }
  for (int i = tid; i < KB2 * NT2 * 512; i += BLOCK) {
    int frag = i >> 9, rem = i & 511, lane = rem >> 4, e = rem & 15;
    int kb = frag / NT2, nt = frag % NT2;
    int n = nt * 16 + (lane & 15);
    int k = kb * 32 + ((lane >> 4) << 4) + e;
    float v = (k < H1_SZ && n < H2_SZ) ? W2[k * H2_SZ + n] : 0.f;
    store_bf1(&sW2f[i], v);
  }
  if (tid < H2_SZ) sW3[tid] = W3[tid];
  if (tid < H1_SZ) sB1[tid] = b1[tid];
  if (tid < N2P)   sB2[tid] = (tid < H2_SZ) ? b2[tid] : 0.f;
  if (tid < E_SZ) { sQ[tid] = queries[(size_t)b * E_SZ + tid] + pos_q[tid]; sOut[tid] = 0.f; }
  __syncthreads();

  const int wave = tid >> 5;
  const int lane = tid & 31;
  unsigned short* attW = &sAtt[wave * 16 * F_SZ];
  unsigned short* haW  = &sHa [wave * 16 * K2P];
  unsigned short* hbW  = &sHb [wave * 16 * N2P];
  float*          wvW  = &sWv [wave * 16];

  // zero the layer-1 K padding (cols 80..95) once — never written again
  for (int i = lane; i < 16 * 16; i += 32)
    haW[(i >> 4) * K2P + 80 + (i & 15)] = 0;

  // q section of the att tile (cols 0..63) is tile-invariant: write once per wave
  #pragma unroll
  for (int it = 0; it < 8; ++it) {
    int idx = lane + it * 32;           // 0..255: (row, 4-float group)
    int r = idx >> 4, g = idx & 15;
    float4 qv = *(const float4*)(sQ + g * 4);
    store_bf4(attW + r * F_SZ + g * 4, qv.x, qv.y, qv.z, qv.w);
  }

  const float bias3 = b3[0];
  float pout0 = 0.f, pout1 = 0.f;   // lane owns output dims e = 2*lane, 2*lane+1

  for (int mt = wave; mt < MTILES; mt += WAVES) {
    const int t0 = mt * 16;

    // prefetch next M-tile's keys rows into cache (global_prefetch_b8)
    {
      int tn = t0 + WAVES * 16 + (lane >> 1);           // 16 rows, 2 lanes per row
      if (tn < T_SZ)
        __builtin_prefetch(keys + ((size_t)b * T_SZ + tn) * E_SZ + (lane & 1) * 32, 0, 3);
    }

    // ---- build att_in tile cols [64..256): [k | q-k | q*k] ----
    #pragma unroll
    for (int it = 0; it < 8; ++it) {
      int idx = lane + it * 32;           // 0..255: (row, 4-float group)
      int r = idx >> 4;
      int g = idx & 15;
      int t = t0 + r;
      float4 kv = make_float4(0.f, 0.f, 0.f, 0.f);
      if (t < T_SZ) {
        float4 kk = *(const float4*)(keys + ((size_t)b * T_SZ + t) * E_SZ + g * 4);
        float4 pk = *(const float4*)(pos_k + (size_t)t * E_SZ + g * 4);
        kv = make_float4(kk.x + pk.x, kk.y + pk.y, kk.z + pk.z, kk.w + pk.w);
      }
      float4 qv = *(const float4*)(sQ + g * 4);
      unsigned short* rowp = attW + r * F_SZ + g * 4;
      store_bf4(rowp + 64,   kv.x,        kv.y,        kv.z,        kv.w);
      store_bf4(rowp + 128,  qv.x - kv.x, qv.y - kv.y, qv.z - kv.z, qv.w - kv.w);
      store_bf4(rowp + 192,  qv.x * kv.x, qv.y * kv.y, qv.z * kv.z, qv.w * kv.w);
    }

    // ---- layer 1: [16x256] @ [256x80] (+b1), 40 WMMAs ----
    v8f acc[NT1];
    #pragma unroll
    for (int nt = 0; nt < NT1; ++nt) {
      float bv = sB1[nt * 16 + (lane & 15)];
      #pragma unroll
      for (int j = 0; j < 8; ++j) acc[nt][j] = bv;
    }
    #pragma unroll
    for (int kb = 0; kb < KB1; ++kb) {
      v16bf a = load_a_frag(attW, F_SZ, kb * 32, lane);
      #pragma unroll
      for (int nt = 0; nt < NT1; ++nt) {
        v16bf bfr = load_b_frag(&sW1f[(kb * NT1 + nt) * 512 + lane * 16]);
        acc[nt] = __builtin_amdgcn_wmma_f32_16x16x32_bf16(
            false, a, false, bfr, (short)0, acc[nt], false, false);
      }
    }
    // ReLU, C-layout -> row-major bf16 [16 x 96] (A-layout staging)
    {
      const int mrow = (lane >> 4) << 3;   // C layout: M = j + 8*(lane/16)
      const int ncol = lane & 15;          //           N = lane % 16
      #pragma unroll
      for (int nt = 0; nt < NT1; ++nt)
        #pragma unroll
        for (int j = 0; j < 8; ++j) {
          float v = acc[nt][j]; v = v > 0.f ? v : 0.f;
          store_bf1(&haW[(mrow + j) * K2P + nt * 16 + ncol], v);
        }
    }

    // ---- layer 2: [16x96] @ [96x48] (+b2), 9 WMMAs ----
    v8f acc2[NT2];
    #pragma unroll
    for (int nt = 0; nt < NT2; ++nt) {
      float bv = sB2[nt * 16 + (lane & 15)];
      #pragma unroll
      for (int j = 0; j < 8; ++j) acc2[nt][j] = bv;
    }
    #pragma unroll
    for (int kb = 0; kb < KB2; ++kb) {
      v16bf a = load_a_frag(haW, K2P, kb * 32, lane);
      #pragma unroll
      for (int nt = 0; nt < NT2; ++nt) {
        v16bf bfr = load_b_frag(&sW2f[(kb * NT2 + nt) * 512 + lane * 16]);
        acc2[nt] = __builtin_amdgcn_wmma_f32_16x16x32_bf16(
            false, a, false, bfr, (short)0, acc2[nt], false, false);
      }
    }
    {
      const int mrow = (lane >> 4) << 3;
      const int ncol = lane & 15;
      #pragma unroll
      for (int nt = 0; nt < NT2; ++nt)
        #pragma unroll
        for (int j = 0; j < 8; ++j) {
          float v = acc2[nt][j]; v = v > 0.f ? v : 0.f;
          store_bf1(&hbW[(mrow + j) * N2P + nt * 16 + ncol], v);
        }
    }

    // ---- layer 3 (40-dot) + sigmoid + mask, lanes 0..15 own one row each ----
    if (lane < 16) {
      int t = t0 + lane;
      float s = bias3;
      #pragma unroll
      for (int j = 0; j < H2_SZ; ++j)
        s += load_bf1(&hbW[lane * N2P + j]) * sW3[j];
      float sc = 1.f / (1.f + __expf(-s));
      float m  = (t < T_SZ) ? (float)key_masks[(size_t)b * T_SZ + t] : 0.f;
      wvW[lane] = sc * m;
    }

    // ---- weighted sum of (keys+pos_k): packed pair reads from att tile ----
    #pragma unroll
    for (int r = 0; r < 16; ++r) {
      float wr = wvW[r];
      bf2v kv2 = *(const bf2v*)(attW + r * F_SZ + 64 + 2 * lane);
      pout0 += wr * (float)kv2.x;
      pout1 += wr * (float)kv2.y;
    }
  }

  atomicAdd(&sOut[2 * lane],     pout0);
  atomicAdd(&sOut[2 * lane + 1], pout1);
  __syncthreads();
  if (tid < E_SZ) out[(size_t)b * E_SZ + tid] = sOut[tid];
}

extern "C" void kernel_launch(void* const* d_in, const int* in_sizes, int n_in,
                              void* d_out, int out_size, void* d_ws, size_t ws_size,
                              hipStream_t stream) {
  (void)in_sizes; (void)n_in; (void)out_size; (void)d_ws; (void)ws_size;
  const float* queries   = (const float*)d_in[0];
  const float* keys      = (const float*)d_in[1];
  const int*   key_masks = (const int*)  d_in[2];
  const float* pos_q     = (const float*)d_in[3];
  const float* pos_k     = (const float*)d_in[4];
  const float* W1 = (const float*)d_in[5];
  const float* b1 = (const float*)d_in[6];
  const float* W2 = (const float*)d_in[7];
  const float* b2 = (const float*)d_in[8];
  const float* W3 = (const float*)d_in[9];
  const float* b3 = (const float*)d_in[10];

  attnpool_wmma_kernel<<<dim3(B_SZ), dim3(BLOCK), 0, stream>>>(
      queries, keys, key_masks, pos_q, pos_k, W1, b1, W2, b2, W3, b3, (float*)d_out);
}
